// QwenDecoderLayer_50964081935159
// MI455X (gfx1250) — matrix operand order repaired and checked by reading, not compiled, not run on hardware
//
#include <hip/hip_runtime.h>
#include <hip/hip_bf16.h>

// ---------------- problem constants (match reference) ----------------
constexpr int BB   = 2;
constexpr int SS   = 1024;
constexpr int HH   = 1024;
constexpr int NHEAD = 16;
constexpr int NKVH  = 4;
constexpr int HDIM  = 64;
constexpr int NEXP  = 64;
constexpr int TOPK  = 8;
constexpr int IEXP  = 512;
constexpr int ISH   = 1408;
constexpr int TT    = BB * SS;          // 2048 tokens
constexpr float EPSV  = 1e-6f;
constexpr float THETA = 1000000.0f;

typedef _Float16 v16h __attribute__((ext_vector_type(16)));
typedef float    v8f  __attribute__((ext_vector_type(8)));

// ---------------- WMMA helpers (CDNA5 wave32) ----------------
__device__ __forceinline__ v8f wmma_f16(v16h a, v16h b, v8f c) {
  // D = A(16x32 f16) * B(32x16 f16) + C(16x16 f32)
  return __builtin_amdgcn_wmma_f32_16x16x32_f16(false, a, false, b, (short)0, c,
                                                false, false);
}

// A-tile (16x32) from a per-lane row pointer (row already selected by caller).
// A layout: lane L holds row M=L%16; elem e -> K = 16*hi + e, hi = L/16.
__device__ __forceinline__ v16h load_a_row(const float* __restrict__ arow,
                                           int k0, int hi) {
  v16h r;
  const float* p0 = arow + k0 + 16 * hi;
  const float* p1 = p0 + 8;
#pragma unroll
  for (int i = 0; i < 8; ++i) r[i] = (_Float16)p0[i];
#pragma unroll
  for (int i = 0; i < 8; ++i) r[8 + i] = (_Float16)p1[i];
  return r;
}

// B-tile (32x16) from torch-layout weight W[N,K] (row-major in K): B[k][n]=W[n][k]
// B layout: lane L holds column N=L%16; elem e -> K = e + 16*(L/16).
__device__ __forceinline__ v16h load_bT_tile(const float* __restrict__ W, int ldw,
                                             int n0, int k0, int lane) {
  int n = n0 + (lane & 15);
  const float* p = W + (size_t)n * ldw + k0 + 16 * (lane >> 4);
  v16h r;
#pragma unroll
  for (int i = 0; i < 16; ++i) r[i] = (_Float16)p[i];
  return r;
}

// B-tile (32x16) from K-major weight W[K,N]: B[k][n]=W[k*ldn+n]
__device__ __forceinline__ v16h load_b_tile(const float* __restrict__ W, int ldn,
                                            int n0, int k0, int lane) {
  int n = n0 + (lane & 15);
  const float* p = W + (size_t)(k0 + 16 * (lane >> 4)) * ldn + n;
  v16h r;
#pragma unroll
  for (int i = 0; i < 16; ++i) r[i] = (_Float16)p[(size_t)i * ldn];
  return r;
}

// ---------------- async global->LDS staging (CDNA5 ASYNCcnt path) ----------------
// 16B cooperative copy: per-lane LDS byte offset in a VGPR, 64-bit global address.
__device__ __forceinline__ void async_copy_b128(void* lds_dst, const void* gsrc) {
  unsigned int loff = (unsigned int)(uintptr_t)lds_dst;         // addr[31:0] = LDS offset
  unsigned long long ga = (unsigned long long)(uintptr_t)gsrc;  // global address
  asm volatile("global_load_async_to_lds_b128 %0, %1, off"
               :: "v"(loff), "v"(ga)
               : "memory");
}
__device__ __forceinline__ void async_wait0() {
  asm volatile("s_wait_asynccnt 0" ::: "memory");
}

// B-tile (32x16) from an LDS slab staged as [32][64+PAD] fp32 (K-major, 64 cols)
constexpr int SLAB_PAD = 4;  // floats; row stride = 68 floats = 272B (16B aligned)
constexpr int SLAB_LD  = 64 + SLAB_PAD;
__device__ __forceinline__ v16h lds_b_tile(const float* s, int n0l, int lane) {
  int n = n0l + (lane & 15);
  const float* p = s + (size_t)(16 * (lane >> 4)) * SLAB_LD + n;
  v16h r;
#pragma unroll
  for (int i = 0; i < 16; ++i) r[i] = (_Float16)p[(size_t)i * SLAB_LD];
  return r;
}

__device__ __forceinline__ float rmax16(float v) {
  v = fmaxf(v, __shfl_xor(v, 1, 32));
  v = fmaxf(v, __shfl_xor(v, 2, 32));
  v = fmaxf(v, __shfl_xor(v, 4, 32));
  v = fmaxf(v, __shfl_xor(v, 8, 32));
  return v;
}
__device__ __forceinline__ float rsum16(float v) {
  v += __shfl_xor(v, 1, 32);
  v += __shfl_xor(v, 2, 32);
  v += __shfl_xor(v, 4, 32);
  v += __shfl_xor(v, 8, 32);
  return v;
}

// ---------------- elementwise kernels ----------------
__global__ __launch_bounds__(256) void rmsnorm_kernel(const float* __restrict__ x,
                                                      const float* __restrict__ w,
                                                      float* __restrict__ out) {
  int t = blockIdx.x;
  __shared__ float red[256];
  const float4* xr = (const float4*)(x + (size_t)t * HH);
  float4 v = xr[threadIdx.x];
  red[threadIdx.x] = v.x * v.x + v.y * v.y + v.z * v.z + v.w * v.w;
  __syncthreads();
  for (int o = 128; o > 0; o >>= 1) {
    if (threadIdx.x < o) red[threadIdx.x] += red[threadIdx.x + o];
    __syncthreads();
  }
  float r = rsqrtf(red[0] / (float)HH + EPSV);
  float4 ww = ((const float4*)w)[threadIdx.x];
  float4 o4;
  o4.x = v.x * r * ww.x; o4.y = v.y * r * ww.y;
  o4.z = v.z * r * ww.z; o4.w = v.w * r * ww.w;
  ((float4*)(out + (size_t)t * HH))[threadIdx.x] = o4;
}

__global__ void zero_kernel(float* __restrict__ p, int n) {
  int i = blockIdx.x * blockDim.x + threadIdx.x;
  if (i < n) p[i] = 0.f;
}

__global__ void add_kernel(const float* __restrict__ a, const float* __restrict__ b,
                           float* __restrict__ out, int n) {
  int i = blockIdx.x * blockDim.x + threadIdx.x;
  if (i < n) out[i] = a[i] + b[i];
}

// in-place RoPE on q [TT, NHEAD*HDIM] and k [TT, NKVH*HDIM]
__global__ __launch_bounds__(256) void rope_kernel(float* __restrict__ q,
                                                   float* __restrict__ k) {
  int t = blockIdx.x;
  int s = t % SS;
  for (int p = threadIdx.x; p < (NHEAD + NKVH) * 32; p += 256) {
    float* base;
    int j;
    if (p < NHEAD * 32) { base = q + (size_t)t * (NHEAD * HDIM) + (p >> 5) * HDIM; j = p & 31; }
    else { int pp = p - NHEAD * 32; base = k + (size_t)t * (NKVH * HDIM) + (pp >> 5) * HDIM; j = pp & 31; }
    float invf = __powf(THETA, -(float)j / 32.0f);
    float fr = (float)s * invf;
    float c = __cosf(fr), si = __sinf(fr);
    float x1 = base[j], x2 = base[j + 32];
    base[j]      = x1 * c - x2 * si;
    base[j + 32] = x2 * c + x1 * si;
  }
}

// ---------------- GEMM kernels (WMMA f16, fp32 accumulate) ----------------
// out[T,N] = X[T,K] @ W^T + bias ; W is torch [N,K]
__global__ __launch_bounds__(128) void gemm_bias_kernel(const float* __restrict__ X,
    const float* __restrict__ W, const float* __restrict__ bias,
    float* __restrict__ out, int K, int N) {
  int wave = threadIdx.x >> 5, lane = threadIdx.x & 31;
  int hi = lane >> 4, lo = lane & 15;
  int m0 = blockIdx.x * 64 + wave * 16;
  int n0 = blockIdx.y * 64;
  const float* arow = X + (size_t)(m0 + lo) * K;
  v8f acc[4] = {};
  for (int k0 = 0; k0 < K; k0 += 32) {
    v16h a = load_a_row(arow, k0, hi);
#pragma unroll
    for (int nt = 0; nt < 4; ++nt)
      acc[nt] = wmma_f16(a, load_bT_tile(W, K, n0 + 16 * nt, k0, lane), acc[nt]);
  }
#pragma unroll
  for (int nt = 0; nt < 4; ++nt) {
    int n = n0 + 16 * nt + lo;
    float bv = bias ? bias[n] : 0.f;
#pragma unroll
    for (int r = 0; r < 8; ++r)
      out[(size_t)(m0 + r + 8 * hi) * N + n] = acc[nt][r] + bv;
  }
}

// out[T,N] = res[T,N] + X[T,K] @ W^T
__global__ __launch_bounds__(128) void gemm_res_kernel(const float* __restrict__ X,
    const float* __restrict__ W, const float* __restrict__ res,
    float* __restrict__ out, int K, int N) {
  int wave = threadIdx.x >> 5, lane = threadIdx.x & 31;
  int hi = lane >> 4, lo = lane & 15;
  int m0 = blockIdx.x * 64 + wave * 16;
  int n0 = blockIdx.y * 64;
  const float* arow = X + (size_t)(m0 + lo) * K;
  v8f acc[4] = {};
  for (int k0 = 0; k0 < K; k0 += 32) {
    v16h a = load_a_row(arow, k0, hi);
#pragma unroll
    for (int nt = 0; nt < 4; ++nt)
      acc[nt] = wmma_f16(a, load_bT_tile(W, K, n0 + 16 * nt, k0, lane), acc[nt]);
  }
#pragma unroll
  for (int nt = 0; nt < 4; ++nt) {
    int n = n0 + 16 * nt + lo;
#pragma unroll
    for (int r = 0; r < 8; ++r) {
      size_t idx = (size_t)(m0 + r + 8 * hi) * N + n;
      out[idx] = res[idx] + acc[nt][r];
    }
  }
}

// shared expert gate/up: out[T,N] = silu(X@Wg^T) * (X@Wu^T) ; Wg,Wu torch [N,K]
__global__ __launch_bounds__(128) void gemm_silu_wt_kernel(const float* __restrict__ X,
    const float* __restrict__ Wg, const float* __restrict__ Wu,
    float* __restrict__ out, int K, int N) {
  int wave = threadIdx.x >> 5, lane = threadIdx.x & 31;
  int hi = lane >> 4, lo = lane & 15;
  int m0 = blockIdx.x * 64 + wave * 16;
  int n0 = blockIdx.y * 64;
  const float* arow = X + (size_t)(m0 + lo) * K;
  v8f ag[4] = {}, au[4] = {};
  for (int k0 = 0; k0 < K; k0 += 32) {
    v16h a = load_a_row(arow, k0, hi);
#pragma unroll
    for (int nt = 0; nt < 4; ++nt) {
      ag[nt] = wmma_f16(a, load_bT_tile(Wg, K, n0 + 16 * nt, k0, lane), ag[nt]);
      au[nt] = wmma_f16(a, load_bT_tile(Wu, K, n0 + 16 * nt, k0, lane), au[nt]);
    }
  }
#pragma unroll
  for (int nt = 0; nt < 4; ++nt) {
    int n = n0 + 16 * nt + lo;
#pragma unroll
    for (int r = 0; r < 8; ++r) {
      float g = ag[nt][r], u = au[nt][r];
      out[(size_t)(m0 + r + 8 * hi) * N + n] = (g / (1.f + __expf(-g))) * u;
    }
  }
}

// shared expert down: moe[t,n] += sig[t] * (X@W^T) ; W torch [N,K]
__global__ __launch_bounds__(128) void gemm_scale_atomic_kernel(const float* __restrict__ X,
    const float* __restrict__ W, const float* __restrict__ sig,
    float* __restrict__ moe, int K, int N) {
  int wave = threadIdx.x >> 5, lane = threadIdx.x & 31;
  int hi = lane >> 4, lo = lane & 15;
  int m0 = blockIdx.x * 64 + wave * 16;
  int n0 = blockIdx.y * 64;
  const float* arow = X + (size_t)(m0 + lo) * K;
  v8f acc[4] = {};
  for (int k0 = 0; k0 < K; k0 += 32) {
    v16h a = load_a_row(arow, k0, hi);
#pragma unroll
    for (int nt = 0; nt < 4; ++nt)
      acc[nt] = wmma_f16(a, load_bT_tile(W, K, n0 + 16 * nt, k0, lane), acc[nt]);
  }
#pragma unroll
  for (int nt = 0; nt < 4; ++nt) {
    int n = n0 + 16 * nt + lo;
#pragma unroll
    for (int r = 0; r < 8; ++r) {
      int t = m0 + r + 8 * hi;
      atomicAdd(&moe[(size_t)t * N + n], sig[t] * acc[nt][r]);
    }
  }
}

// ---------------- attention: flash-style, WMMA for QK^T and P*V ----------------
__global__ __launch_bounds__(128) void attn_kernel(const float* __restrict__ qb,
    const float* __restrict__ kb, const float* __restrict__ vb,
    float* __restrict__ ob) {
  int h = blockIdx.y, bidx = blockIdx.z;
  int wave = threadIdx.x >> 5, lane = threadIdx.x & 31;
  int hi = lane >> 4, lo = lane & 15;
  int kh = h >> 2;                       // GQA: 4 q-heads per kv-head
  int qw = blockIdx.x * 64 + wave * 16;  // first q row of this wave
  __shared__ _Float16 pbuf[4][16][32];

  const float* qrow = qb + ((size_t)(bidx * SS) + qw + lo) * (NHEAD * HDIM) + h * HDIM;
  v16h aq0 = load_a_row(qrow, 0, hi);
  v16h aq1 = load_a_row(qrow, 32, hi);
#pragma unroll
  for (int i = 0; i < 16; ++i) {  // fold 1/sqrt(64) into Q
    aq0[i] = aq0[i] * (_Float16)0.125f;
    aq1[i] = aq1[i] * (_Float16)0.125f;
  }

  v8f o0 = {}, o1 = {}, o2 = {}, o3 = {};
  float m[8], l[8];
#pragma unroll
  for (int r = 0; r < 8; ++r) { m[r] = -1e30f; l[r] = 0.f; }

  const float* Kb = kb + (size_t)(bidx * SS) * (NKVH * HDIM) + kh * HDIM;
  int qhi = qw + 15;
  for (int j0 = 0; j0 <= qhi; j0 += 32) {
    v8f s0 = {}, s1 = {};
    s0 = wmma_f16(aq0, load_bT_tile(Kb, NKVH * HDIM, j0, 0, lane), s0);
    s0 = wmma_f16(aq1, load_bT_tile(Kb, NKVH * HDIM, j0, 32, lane), s0);
    s1 = wmma_f16(aq0, load_bT_tile(Kb, NKVH * HDIM, j0 + 16, 0, lane), s1);
    s1 = wmma_f16(aq1, load_bT_tile(Kb, NKVH * HDIM, j0 + 16, 32, lane), s1);
#pragma unroll
    for (int r = 0; r < 8; ++r) {  // causal mask
      int qr = qw + r + 8 * hi;
      if (j0 + lo > qr) s0[r] = -1e30f;
      if (j0 + 16 + lo > qr) s1[r] = -1e30f;
    }
#pragma unroll
    for (int r = 0; r < 8; ++r) {  // online softmax per row
      float nm = rmax16(fmaxf(s0[r], s1[r]));
      nm = fmaxf(m[r], nm);
      float alpha = __expf(m[r] - nm);
      float p0 = __expf(s0[r] - nm);
      float p1 = __expf(s1[r] - nm);
      l[r] = l[r] * alpha + rsum16(p0 + p1);
      m[r] = nm;
      o0[r] *= alpha; o1[r] *= alpha; o2[r] *= alpha; o3[r] *= alpha;
      pbuf[wave][r + 8 * hi][lo] = (_Float16)p0;
      pbuf[wave][r + 8 * hi][16 + lo] = (_Float16)p1;
    }
    // re-layout P (C-layout in LDS) into WMMA-A layout (wave-synchronous; DS in-order)
    v16h ap;
    {
      const _Float16* pr = &pbuf[wave][lo][0];
#pragma unroll
      for (int i = 0; i < 8; ++i) ap[i] = pr[16 * hi + i];
#pragma unroll
      for (int i = 0; i < 8; ++i) ap[8 + i] = pr[16 * hi + 8 + i];
    }
    const float* Vb = vb + (size_t)(bidx * SS + j0) * (NKVH * HDIM) + kh * HDIM;
    o0 = wmma_f16(ap, load_b_tile(Vb, NKVH * HDIM, 0, 0, lane), o0);
    o1 = wmma_f16(ap, load_b_tile(Vb, NKVH * HDIM, 16, 0, lane), o1);
    o2 = wmma_f16(ap, load_b_tile(Vb, NKVH * HDIM, 32, 0, lane), o2);
    o3 = wmma_f16(ap, load_b_tile(Vb, NKVH * HDIM, 48, 0, lane), o3);
  }
#pragma unroll
  for (int r = 0; r < 8; ++r) {
    int srow = qw + r + 8 * hi;
    float invl = 1.f / l[r];
    float* op = ob + ((size_t)(bidx * SS) + srow) * (NHEAD * HDIM) + h * HDIM;
    op[lo] = o0[r] * invl;
    op[16 + lo] = o1[r] * invl;
    op[32 + lo] = o2[r] * invl;
    op[48 + lo] = o3[r] * invl;
  }
}

// ---------------- MoE routing ----------------
__global__ __launch_bounds__(64) void router_kernel(const float* __restrict__ y,
    const float* __restrict__ gw, const float* __restrict__ shg,
    int* __restrict__ topi, float* __restrict__ topw, float* __restrict__ sig) {
  int t = blockIdx.x, e = threadIdx.x;
  __shared__ float logits[NEXP];
  __shared__ float parts[NEXP];
  const float* yr = y + (size_t)t * HH;
  const float4* y4 = (const float4*)yr;
  const float4* w4 = (const float4*)(gw + (size_t)e * HH);
  float acc = 0.f;
  for (int k = 0; k < HH / 4; ++k) {
    float4 a = y4[k], b = w4[k];
    acc += a.x * b.x + a.y * b.y + a.z * b.z + a.w * b.w;
  }
  logits[e] = acc;
  float p = 0.f;
  for (int k = e * 16; k < e * 16 + 16; ++k) p += yr[k] * shg[k];
  parts[e] = p;
  __syncthreads();
  if (e == 0) {
    float mx = logits[0];
    for (int i = 1; i < NEXP; ++i) mx = fmaxf(mx, logits[i]);
    float sm = 0.f;
    for (int i = 0; i < NEXP; ++i) { logits[i] = __expf(logits[i] - mx); sm += logits[i]; }
    float inv = 1.f / sm;
    float wsum = 0.f;
    for (int s = 0; s < TOPK; ++s) {  // deterministic first-max tie-break
      int bi = 0; float bv = -1.f;
      for (int i = 0; i < NEXP; ++i) { float v = logits[i]; if (v > bv) { bv = v; bi = i; } }
      topi[t * TOPK + s] = bi;
      topw[t * TOPK + s] = bv * inv;
      wsum += bv * inv;
      logits[bi] = -2.f;
    }
    float invw = 1.f / wsum;
    for (int s = 0; s < TOPK; ++s) topw[t * TOPK + s] *= invw;
    float d = 0.f;
    for (int i = 0; i < NEXP; ++i) d += parts[i];
    sig[t] = 1.f / (1.f + __expf(-d));
  }
}

// deterministic per-expert token lists (one thread per expert, token order)
__global__ __launch_bounds__(64) void build_lists_kernel(const int* __restrict__ topi,
    int* __restrict__ list, int* __restrict__ counts, int* __restrict__ offs) {
  int e = threadIdx.x;
  int c = 0;
  for (int i = 0; i < TT * TOPK; ++i)
    if (topi[i] == e) list[e * TT + (c++)] = i;  // i packs t*8+s
  counts[e] = c;
  __syncthreads();
  if (e == 0) {
    int o = 0;
    for (int k = 0; k < NEXP; ++k) { offs[k] = o; o += counts[k]; }
    offs[NEXP] = o;
  }
}

// expert gate/up over gathered rows: inter[offs[e]+i, :] = silu(y@wg[e]) * (y@wu[e])
// B panels (32x64 fp32 slabs of w_gate/w_up) are double-buffered in LDS via
// global_load_async_to_lds_b128 (ASYNCcnt), shared by all 4 waves of the block.
__global__ __launch_bounds__(128) void expert_gateup_kernel(const float* __restrict__ y,
    const float* __restrict__ wg_all, const float* __restrict__ wu_all,
    const int* __restrict__ list, const int* __restrict__ counts,
    const int* __restrict__ offs, float* __restrict__ inter) {
  __shared__ float swg[2][32][SLAB_LD];
  __shared__ float swu[2][32][SLAB_LD];
  int e = blockIdx.x;
  int cnt = counts[e];
  if (cnt == 0 || blockIdx.z * 64 >= cnt) return;  // block-uniform exit
  int wave = threadIdx.x >> 5, lane = threadIdx.x & 31;
  int m0 = blockIdx.z * 64 + wave * 16;
  bool active = (m0 < cnt);
  int hi = lane >> 4, lo = lane & 15;
  int n0 = blockIdx.y * 64;
  int row = m0 + lo;
  int tok = list[e * TT + ((active && row < cnt) ? row : 0)] >> 3;
  const float* arow = y + (size_t)tok * HH;
  const float* wg = wg_all + (size_t)e * HH * IEXP + n0;  // 64-col panel base
  const float* wu = wu_all + (size_t)e * HH * IEXP + n0;

  auto stage = [&](int buf, int k0) {  // copy 32x64 fp32 slab of each weight
#pragma unroll
    for (int i = 0; i < 4; ++i) {
      int c = threadIdx.x + i * 128;    // 0..511 chunks of 16B
      int kr = c >> 4, cc = (c & 15) * 4;
      async_copy_b128(&swg[buf][kr][cc], wg + (size_t)(k0 + kr) * IEXP + cc);
      async_copy_b128(&swu[buf][kr][cc], wu + (size_t)(k0 + kr) * IEXP + cc);
    }
  };

  stage(0, 0);
  async_wait0();
  __syncthreads();

  v8f ag[4] = {}, au[4] = {};
  constexpr int NSLAB = HH / 32;
  for (int ks = 0; ks < NSLAB; ++ks) {
    int buf = ks & 1;
    if (ks + 1 < NSLAB) stage(buf ^ 1, (ks + 1) * 32);
    v16h a = load_a_row(arow, ks * 32, hi);
#pragma unroll
    for (int nt = 0; nt < 4; ++nt) {
      ag[nt] = wmma_f16(a, lds_b_tile(&swg[buf][0][0], nt * 16, lane), ag[nt]);
      au[nt] = wmma_f16(a, lds_b_tile(&swu[buf][0][0], nt * 16, lane), au[nt]);
    }
    async_wait0();
    __syncthreads();
  }

  if (active) {
    int obase = offs[e];
#pragma unroll
    for (int nt = 0; nt < 4; ++nt) {
#pragma unroll
      for (int r = 0; r < 8; ++r) {
        int rr = m0 + r + 8 * hi;
        if (rr < cnt) {
          float g = ag[nt][r], u = au[nt][r];
          inter[(size_t)(obase + rr) * IEXP + n0 + 16 * nt + lo] =
              (g / (1.f + __expf(-g))) * u;
        }
      }
    }
  }
}

// expert down: moe[t,:] += cw * (inter_rows @ w_down[e]) ; async-LDS staged B panel
__global__ __launch_bounds__(128) void expert_down_kernel(const float* __restrict__ inter,
    const float* __restrict__ wd_all, const int* __restrict__ list,
    const int* __restrict__ counts, const int* __restrict__ offs,
    const float* __restrict__ topw, float* __restrict__ moe) {
  __shared__ float swd[2][32][SLAB_LD];
  int e = blockIdx.x;
  int cnt = counts[e];
  if (cnt == 0 || blockIdx.z * 64 >= cnt) return;  // block-uniform exit
  int wave = threadIdx.x >> 5, lane = threadIdx.x & 31;
  int m0 = blockIdx.z * 64 + wave * 16;
  bool active = (m0 < cnt);
  int hi = lane >> 4, lo = lane & 15;
  int n0 = blockIdx.y * 64;
  int obase = offs[e];
  int row = m0 + lo;
  if (!active || row >= cnt) row = 0;
  const float* arow = inter + (size_t)(obase + ((row < cnt) ? row : 0)) * IEXP;
  const float* wd = wd_all + (size_t)e * IEXP * HH + n0;  // 64-col panel base

  auto stage = [&](int buf, int k0) {
#pragma unroll
    for (int i = 0; i < 4; ++i) {
      int c = threadIdx.x + i * 128;
      int kr = c >> 4, cc = (c & 15) * 4;
      async_copy_b128(&swd[buf][kr][cc], wd + (size_t)(k0 + kr) * HH + cc);
    }
  };

  stage(0, 0);
  async_wait0();
  __syncthreads();

  v8f acc[4] = {};
  constexpr int NSLAB = IEXP / 32;
  for (int ks = 0; ks < NSLAB; ++ks) {
    int buf = ks & 1;
    if (ks + 1 < NSLAB) stage(buf ^ 1, (ks + 1) * 32);
    v16h a = load_a_row(arow, ks * 32, hi);
#pragma unroll
    for (int nt = 0; nt < 4; ++nt)
      acc[nt] = wmma_f16(a, lds_b_tile(&swd[buf][0][0], nt * 16, lane), acc[nt]);
    async_wait0();
    __syncthreads();
  }

  if (active) {
#pragma unroll
    for (int nt = 0; nt < 4; ++nt) {
      int n = n0 + 16 * nt + lo;
#pragma unroll
      for (int r = 0; r < 8; ++r) {
        int rr = m0 + r + 8 * hi;
        if (rr < cnt) {
          int packed = list[e * TT + rr];
          int t = packed >> 3;
          atomicAdd(&moe[(size_t)t * HH + n], topw[packed] * acc[nt][r]);
        }
      }
    }
  }
}

// ---------------- launcher ----------------
extern "C" void kernel_launch(void* const* d_in, const int* in_sizes, int n_in,
                              void* d_out, int out_size, void* d_ws, size_t ws_size,
                              hipStream_t stream) {
  const float* hidden = (const float*)d_in[0];
  const float* ln1_w  = (const float*)d_in[1];
  const float* ln2_w  = (const float*)d_in[2];
  const float* q_w    = (const float*)d_in[3];
  const float* q_b    = (const float*)d_in[4];
  const float* k_w    = (const float*)d_in[5];
  const float* k_b    = (const float*)d_in[6];
  const float* v_w    = (const float*)d_in[7];
  const float* v_b    = (const float*)d_in[8];
  const float* o_w    = (const float*)d_in[9];
  const float* gate_w = (const float*)d_in[10];
  const float* w_gate = (const float*)d_in[11];
  const float* w_up   = (const float*)d_in[12];
  const float* w_down = (const float*)d_in[13];
  const float* s_gate_w = (const float*)d_in[14];
  const float* s_up_w   = (const float*)d_in[15];
  const float* s_down_w = (const float*)d_in[16];
  const float* sh_gate_w = (const float*)d_in[17];
  float* out = (float*)d_out;

  size_t off = 0;
  auto alloc = [&](size_t bytes) -> void* {
    void* p = (char*)d_ws + off;
    off = (off + bytes + 255) & ~(size_t)255;
    return p;
  };
  float* xn   = (float*)alloc((size_t)TT * HH * 4);
  float* qb   = (float*)alloc((size_t)TT * NHEAD * HDIM * 4);
  float* kb   = (float*)alloc((size_t)TT * NKVH * HDIM * 4);
  float* vb   = (float*)alloc((size_t)TT * NKVH * HDIM * 4);
  float* ob   = (float*)alloc((size_t)TT * HH * 4);
  float* hb   = (float*)alloc((size_t)TT * HH * 4);
  float* yb   = (float*)alloc((size_t)TT * HH * 4);
  float* moe  = (float*)alloc((size_t)TT * HH * 4);
  int*   topi = (int*)alloc((size_t)TT * TOPK * 4);
  float* topw = (float*)alloc((size_t)TT * TOPK * 4);
  float* sig  = (float*)alloc((size_t)TT * 4);
  int*   list = (int*)alloc((size_t)NEXP * TT * 4);
  int*   counts = (int*)alloc(NEXP * 4);
  int*   offs   = (int*)alloc((NEXP + 1) * 4);
  float* inter  = (float*)alloc((size_t)TT * TOPK * IEXP * 4);
  float* sinter = (float*)alloc((size_t)TT * ISH * 4);

  // 1. RMSNorm 1
  rmsnorm_kernel<<<TT, 256, 0, stream>>>(hidden, ln1_w, xn);
  // 2. QKV projections (WMMA)
  gemm_bias_kernel<<<dim3(TT / 64, (NHEAD * HDIM) / 64), 128, 0, stream>>>(
      xn, q_w, q_b, qb, HH, NHEAD * HDIM);
  gemm_bias_kernel<<<dim3(TT / 64, (NKVH * HDIM) / 64), 128, 0, stream>>>(
      xn, k_w, k_b, kb, HH, NKVH * HDIM);
  gemm_bias_kernel<<<dim3(TT / 64, (NKVH * HDIM) / 64), 128, 0, stream>>>(
      xn, v_w, v_b, vb, HH, NKVH * HDIM);
  // 3. RoPE
  rope_kernel<<<TT, 256, 0, stream>>>(qb, kb);
  // 4. causal GQA attention (WMMA flash)
  attn_kernel<<<dim3(SS / 64, NHEAD, BB), 128, 0, stream>>>(qb, kb, vb, ob);
  // 5. O-proj + residual
  gemm_res_kernel<<<dim3(TT / 64, HH / 64), 128, 0, stream>>>(ob, o_w, hidden, hb,
                                                              NHEAD * HDIM, HH);
  // 6. RMSNorm 2
  rmsnorm_kernel<<<TT, 256, 0, stream>>>(hb, ln2_w, yb);
  // 7. router + shared-gate sigmoid
  router_kernel<<<TT, 64, 0, stream>>>(yb, gate_w, sh_gate_w, topi, topw, sig);
  // 8. per-expert token lists
  build_lists_kernel<<<1, 64, 0, stream>>>(topi, list, counts, offs);
  // 9. zero MoE accumulator
  zero_kernel<<<(TT * HH + 255) / 256, 256, 0, stream>>>(moe, TT * HH);
  // 10/11. routed experts (gathered WMMA GEMMs, async-LDS weight staging)
  expert_gateup_kernel<<<dim3(NEXP, IEXP / 64, TT / 64), 128, 0, stream>>>(
      yb, w_gate, w_up, list, counts, offs, inter);
  expert_down_kernel<<<dim3(NEXP, HH / 64, TT / 64), 128, 0, stream>>>(
      inter, w_down, list, counts, offs, topw, moe);
  // 12/13. shared expert
  gemm_silu_wt_kernel<<<dim3(TT / 64, ISH / 64), 128, 0, stream>>>(
      yb, s_gate_w, s_up_w, sinter, HH, ISH);
  gemm_scale_atomic_kernel<<<dim3(TT / 64, HH / 64), 128, 0, stream>>>(
      sinter, s_down_w, sig, moe, ISH, HH);
  // 14. final residual
  add_kernel<<<(TT * HH + 255) / 256, 256, 0, stream>>>(hb, moe, out, TT * HH);
  (void)in_sizes; (void)n_in; (void)out_size; (void)ws_size;
}
